// PointcloudTokenizer_72078141162102
// MI455X (gfx1250) — matrix operand order repaired and checked by reading, not compiled, not run on hardware
//
#include <hip/hip_runtime.h>

#define B_   16
#define N_   8192
#define G_   512
#define K_   32
#define TOK_ 384
#define M_   (B_*G_)
#define GPB_ 4                 // groups per MLP block
#define R_   (GPB_*K_)         // 128 rows per block
#define MT_  (R_/16)           // 8 M-tiles per block

typedef __attribute__((ext_vector_type(16))) _Float16 v16h;
typedef __attribute__((ext_vector_type(8)))  _Float16 v8h;
typedef __attribute__((ext_vector_type(8)))  float    v8f;

// ---------------- prep: f32 weights -> f16, fold BN into scale/bias ----------------
__global__ void prep_kernel(const float* __restrict__ w2, const float* __restrict__ w3,
                            const float* __restrict__ w4,
                            const float* __restrict__ g1, const float* __restrict__ be1,
                            const float* __restrict__ mu1, const float* __restrict__ va1,
                            const float* __restrict__ g2, const float* __restrict__ be2,
                            const float* __restrict__ mu2, const float* __restrict__ va2,
                            _Float16* __restrict__ w2h, _Float16* __restrict__ w3h,
                            _Float16* __restrict__ w4h,
                            float* __restrict__ s1, float* __restrict__ b1n,
                            float* __restrict__ s2, float* __restrict__ b2n)
{
  int i = blockIdx.x * 256 + threadIdx.x;
  const int T0 = 256*128, T1 = 512*512, T2 = 384*512;
  if (i < T0) { w2h[i] = (_Float16)w2[i]; return; }
  i -= T0;
  if (i < T1) { w3h[i] = (_Float16)w3[i]; return; }
  i -= T1;
  if (i < T2) { w4h[i] = (_Float16)w4[i]; return; }
  i -= T2;
  if (i < 128) { float s = g1[i] * rsqrtf(va1[i] + 1e-5f); s1[i] = s; b1n[i] = be1[i] - mu1[i]*s; return; }
  i -= 128;
  if (i < 512) { float s = g2[i] * rsqrtf(va2[i] + 1e-5f); s2[i] = s; b2n[i] = be2[i] - mu2[i]*s; }
}

// ---------------- FPS: one block per batch; 512 serial rounds, 2 barriers/round ----------------
__global__ void __launch_bounds__(512) fps_kernel(const float* __restrict__ pts,
                                                  float* __restrict__ centers_ws,
                                                  float* __restrict__ centers_out)
{
  __shared__ float s_pv[16];
  __shared__ int   s_pi[16];
  __shared__ int   s_win;
  const int b = blockIdx.x, t = threadIdx.x;       // 512 threads = 16 waves
  const int lane = t & 31, wv = t >> 5;
  const float* pb = pts + (size_t)b * N_ * 3;

  float px[16], py[16], pz[16], mind[16];
  #pragma unroll
  for (int i = 0; i < 16; ++i) {
    int p = t + i * 512;
    px[i] = pb[p*3+0]; py[i] = pb[p*3+1]; pz[i] = pb[p*3+2];
    mind[i] = 1e10f;
  }
  int lastIdx = 0;
  for (int it = 0; it < G_; ++it) {
    float lx = pb[lastIdx*3+0], ly = pb[lastIdx*3+1], lz = pb[lastIdx*3+2];
    if (t == 0) {
      size_t o = ((size_t)b * G_ + it) * 3;
      centers_ws[o+0] = lx; centers_ws[o+1] = ly; centers_ws[o+2] = lz;
      centers_out[o+0] = lx; centers_out[o+1] = ly; centers_out[o+2] = lz;
    }
    float bv = -1.0f; int bi = 0;
    #pragma unroll
    for (int i = 0; i < 16; ++i) {
      float dx = px[i]-lx, dy = py[i]-ly, dz = pz[i]-lz;
      float d2 = dx*dx + dy*dy + dz*dz;
      float mn = fminf(mind[i], d2);
      mind[i] = mn;
      if (mn > bv) { bv = mn; bi = t + i*512; }
    }
    // wave-level argmax (no barriers)
    for (int off = 16; off > 0; off >>= 1) {
      float ov = __shfl_xor(bv, off); int oi = __shfl_xor(bi, off);
      if (ov > bv || (ov == bv && oi < bi)) { bv = ov; bi = oi; }
    }
    if (lane == 0) { s_pv[wv] = bv; s_pi[wv] = bi; }
    __syncthreads();
    if (wv == 0) {
      float v  = (lane < 16) ? s_pv[lane] : -1.0f;
      int   ii = (lane < 16) ? s_pi[lane] : 0x7fffffff;
      for (int off = 16; off > 0; off >>= 1) {
        float ov = __shfl_xor(v, off); int oi = __shfl_xor(ii, off);
        if (ov > v || (ov == v && oi < ii)) { v = ov; ii = oi; }
      }
      if (lane == 0) s_win = ii;
    }
    __syncthreads();
    lastIdx = s_win;
  }
}

// ---------------- kNN: one block per (b,g); d2 cached in registers, 2 barriers/round ----------------
__global__ void __launch_bounds__(256) knn_kernel(const float* __restrict__ pts,
                                                  const float* __restrict__ centers_ws,
                                                  int* __restrict__ kidx)
{
  __shared__ float s_pv[8];
  __shared__ int   s_pi[8];
  __shared__ int   s_win;
  const int m = blockIdx.x, t = threadIdx.x;       // 256 threads = 8 waves
  const int lane = t & 31, wv = t >> 5;
  const int b = m >> 9;  // / G_
  const float cx = centers_ws[(size_t)m*3+0];
  const float cy = centers_ws[(size_t)m*3+1];
  const float cz = centers_ws[(size_t)m*3+2];
  const float* pb = pts + (size_t)b * N_ * 3;

  float vals[32];
  #pragma unroll
  for (int i = 0; i < 32; ++i) {
    int p = t + i*256;
    float dx = pb[p*3+0]-cx, dy = pb[p*3+1]-cy, dz = pb[p*3+2]-cz;
    vals[i] = dx*dx + dy*dy + dz*dz;
  }
  float lv = 3.4e38f; int li = 0;
  #pragma unroll
  for (int i = 0; i < 32; ++i) if (vals[i] < lv) { lv = vals[i]; li = t + i*256; }

  for (int r = 0; r < K_; ++r) {
    float bv = lv; int bi = li;
    for (int off = 16; off > 0; off >>= 1) {
      float ov = __shfl_xor(bv, off); int oi = __shfl_xor(bi, off);
      if (ov < bv || (ov == bv && oi < bi)) { bv = ov; bi = oi; }
    }
    if (lane == 0) { s_pv[wv] = bv; s_pi[wv] = bi; }
    __syncthreads();
    if (wv == 0) {
      float v  = (lane < 8) ? s_pv[lane] : 3.4e38f;
      int   ii = (lane < 8) ? s_pi[lane] : 0x7fffffff;
      for (int off = 16; off > 0; off >>= 1) {
        float ov = __shfl_xor(v, off); int oi = __shfl_xor(ii, off);
        if (ov < v || (ov == v && oi < ii)) { v = ov; ii = oi; }
      }
      if (lane == 0) s_win = ii;
    }
    __syncthreads();
    int w = s_win;
    if (t == 0) kidx[(size_t)m*K_ + r] = w;
    if ((w & 255) == t) {            // only the winning thread refreshes its cached argmin
      int ws = w >> 8;
      lv = 3.4e38f; li = 0;
      #pragma unroll
      for (int i = 0; i < 32; ++i) {
        if (i == ws) vals[i] = 3.4e38f;
        float v = vals[i];
        if (v < lv) { lv = v; li = t + i*256; }
      }
    }
  }
}

// ---------------- WMMA fragment helpers (ISA 16-bit A / B / C layouts) ----------------
__device__ __forceinline__ v16h load_a_frag(const _Float16* a, int lda, int mtile, int ktile, int lane) {
  int row = mtile*16 + (lane & 15);
  int kb  = ktile*32 + ((lane >> 4) << 3);     // lane<16: K{0-7,16-23}, lane>=16: K{8-15,24-31}
  const _Float16* p = a + row*lda + kb;
  v8h lo = *(const v8h*)(p);
  v8h hi = *(const v8h*)(p + 16);
  v16h r;
  #pragma unroll
  for (int i = 0; i < 8; ++i) { r[i] = lo[i]; r[i+8] = hi[i]; }
  return r;
}
__device__ __forceinline__ v16h load_b_frag(const _Float16* bg, int ldb, int ntile, int ktile, int lane) {
  int col = ntile*16 + (lane & 15);
  int kb  = ktile*32 + ((lane >> 4) << 3);     // lane<16: K{0-7,16-23}, lane>=16: K{8-15,24-31}
  const _Float16* p = bg + (size_t)col*ldb + kb;
  v8h lo = *(const v8h*)(p);
  v8h hi = *(const v8h*)(p + 16);
  v16h r;
  #pragma unroll
  for (int i = 0; i < 8; ++i) { r[i] = lo[i]; r[i+8] = hi[i]; }
  return r;
}
__device__ __forceinline__ float cmax8(v8f c) {
  float pm = c[0];
  #pragma unroll
  for (int r = 1; r < 8; ++r) pm = fmaxf(pm, c[r]);
  return pm;
}

// ---------------- fused per-4-group MLP with 8x B-fragment reuse ----------------
__global__ void __launch_bounds__(256) mlp_kernel(
    const float* __restrict__ pts, const int* __restrict__ kidx,
    const float* __restrict__ centers_ws,
    const float* __restrict__ w1, const float* __restrict__ s1, const float* __restrict__ b1n,
    const _Float16* __restrict__ w2h, const float* __restrict__ b2,
    const _Float16* __restrict__ w3h, const float* __restrict__ s2, const float* __restrict__ b2n,
    const _Float16* __restrict__ w4h, const float* __restrict__ b4,
    float* __restrict__ tokens)
{
  // LDS pool: f1h 128x128 f16 | f2h 128x256 f16 | f3h 128x512 f16 | hbuf 4x512 f32 | s_g 4x256 f32 | sh_x 128x3 f32
  __shared__ __align__(64) char pool[243200];
  _Float16* f1h = (_Float16*)(pool);              // 32768 B
  _Float16* f2h = (_Float16*)(pool + 32768);      // 65536 B
  _Float16* f3h = (_Float16*)(pool + 98304);      // 131072 B
  float* hbuf   = (float*)(pool + 229376);        // 8192 B
  float* s_g    = (float*)(pool + 237568);        // 4096 B
  float* sh_x   = (float*)(pool + 241664);        // 1536 B

  const int m0   = blockIdx.x * GPB_;
  const int tid  = threadIdx.x;
  const int lane = tid & 31;
  const int wv   = tid >> 5;   // 8 waves

  // gather centered points for GPB_ groups: sh_x[row*3+c], row = g*32+k
  for (int o = tid; o < GPB_*K_*3; o += 256) {
    int g = o / 96, rem = o - g*96;
    int k = rem / 3, cc = rem - k*3;
    int m = m0 + g, b = m >> 9;
    int pidx = kidx[(size_t)m*K_ + k];
    sh_x[(g*K_ + k)*3 + cc] = pts[((size_t)b*N_ + pidx)*3 + cc] - centers_ws[(size_t)m*3 + cc];
  }
  __syncthreads();

  // stage 1 (VALU, K-dim=3): relu(bn1(x @ w1^T)) -> f16, 128x128
  for (int o = tid; o < R_*128; o += 256) {
    int row = o >> 7, d = o & 127;
    float acc = sh_x[row*3+0]*w1[d*3+0] + sh_x[row*3+1]*w1[d*3+1] + sh_x[row*3+2]*w1[d*3+2];
    float v = fmaxf(acc * s1[d] + b1n[d], 0.0f);
    f1h[row*128 + d] = (_Float16)v;
  }
  __syncthreads();

  // GEMM2 (WMMA): [128x128]@[128x256] + b2 -> f2h.  1 B-load feeds 8 WMMAs.
  for (int nt = wv; nt < 16; nt += 8) {
    v8f acc[MT_];
    #pragma unroll
    for (int i = 0; i < MT_; ++i) acc[i] = {};
    for (int kt = 0; kt < 4; ++kt) {
      v16h bfr = load_b_frag(w2h, 128, nt, kt, lane);
      #pragma unroll
      for (int mt = 0; mt < MT_; ++mt) {
        v16h afr = load_a_frag(f1h, 128, mt, kt, lane);
        acc[mt] = __builtin_amdgcn_wmma_f32_16x16x32_f16(false, afr, false, bfr, (short)0, acc[mt], false, false);
      }
    }
    int col = nt*16 + (lane & 15);
    float bias = b2[col];
    #pragma unroll
    for (int mt = 0; mt < MT_; ++mt) {
      int row0 = mt*16 + ((lane >> 4) << 3);
      #pragma unroll
      for (int r = 0; r < 8; ++r) f2h[(row0+r)*256 + col] = (_Float16)(acc[mt][r] + bias);
    }
  }
  __syncthreads();

  // g = per-group column max of f2
  for (int e = tid; e < GPB_*256; e += 256) {
    int gg = e >> 8, ee = e & 255;
    float mx = (float)f2h[(gg*K_)*256 + ee];
    for (int k = 1; k < K_; ++k) mx = fmaxf(mx, (float)f2h[(gg*K_ + k)*256 + ee]);
    s_g[gg*256 + ee] = mx;
  }
  __syncthreads();
  // h = W3[:, :256] @ g per group (identical for all 32 rows of a group)
  for (int f = tid; f < GPB_*512; f += 256) {
    int gg = f >> 9, ff = f & 511;
    const _Float16* wr = w3h + (size_t)ff * 512;
    const float* gv = s_g + gg*256;
    float acc = 0.0f;
    for (int e = 0; e < 256; ++e) acc += (float)wr[e] * gv[e];
    hbuf[gg*512 + ff] = acc;
  }
  __syncthreads();

  // GEMM3 (WMMA): [128x256] @ W3[:,256:]^T + h -> relu(bn2) -> f3h
  for (int nt = wv; nt < 32; nt += 8) {
    v8f acc[MT_];
    #pragma unroll
    for (int i = 0; i < MT_; ++i) acc[i] = {};
    for (int kt = 0; kt < 8; ++kt) {
      v16h bfr = load_b_frag(w3h + 256, 512, nt, kt, lane);
      if (kt + 1 < 8) __builtin_prefetch(w3h + 256 + (size_t)(nt*16)*512 + (kt+1)*32, 0, 1);
      #pragma unroll
      for (int mt = 0; mt < MT_; ++mt) {
        v16h afr = load_a_frag(f2h, 256, mt, kt, lane);
        acc[mt] = __builtin_amdgcn_wmma_f32_16x16x32_f16(false, afr, false, bfr, (short)0, acc[mt], false, false);
      }
    }
    int col = nt*16 + (lane & 15);
    float sc = s2[col], bb = b2n[col];
    #pragma unroll
    for (int mt = 0; mt < MT_; ++mt) {
      float hh = hbuf[(mt >> 1)*512 + col];   // group = mt/2
      int row0 = mt*16 + ((lane >> 4) << 3);
      #pragma unroll
      for (int r = 0; r < 8; ++r) {
        float v = fmaxf((acc[mt][r] + hh) * sc + bb, 0.0f);
        f3h[(row0+r)*512 + col] = (_Float16)v;
      }
    }
  }
  __syncthreads();

  // GEMM4 (WMMA): [128x512]@[512x384] -> per-group row-max finalized in registers -> tokens
  for (int nt = wv; nt < 24; nt += 8) {
    v8f acc[MT_];
    #pragma unroll
    for (int i = 0; i < MT_; ++i) acc[i] = {};
    for (int kt = 0; kt < 16; ++kt) {
      v16h bfr = load_b_frag(w4h, 512, nt, kt, lane);
      if (kt + 1 < 16) __builtin_prefetch(w4h + (size_t)(nt*16)*512 + (kt+1)*32, 0, 1);
      #pragma unroll
      for (int mt = 0; mt < MT_; ++mt) {
        v16h afr = load_a_frag(f3h, 512, mt, kt, lane);
        acc[mt] = __builtin_amdgcn_wmma_f32_16x16x32_f16(false, afr, false, bfr, (short)0, acc[mt], false, false);
      }
    }
    int col = nt*16 + (lane & 15);
    float bias = b4[col];
    float pg[GPB_];
    #pragma unroll
    for (int g = 0; g < GPB_; ++g) {
      float p = fmaxf(cmax8(acc[2*g]), cmax8(acc[2*g+1]));   // rows of group g
      p = fmaxf(p, __shfl_xor(p, 16));                       // lanes L/L+16 hold same col
      pg[g] = p;
    }
    if (lane < 16) {
      #pragma unroll
      for (int g = 0; g < GPB_; ++g)
        tokens[(size_t)(m0+g)*TOK_ + col] = pg[g] + bias;
    }
  }
}

// ---------------- launch ----------------
extern "C" void kernel_launch(void* const* d_in, const int* in_sizes, int n_in,
                              void* d_out, int out_size, void* d_ws, size_t ws_size,
                              hipStream_t stream)
{
  const float* pts = (const float*)d_in[0];
  const float* w1  = (const float*)d_in[1];
  const float* g1  = (const float*)d_in[2];
  const float* be1 = (const float*)d_in[3];
  const float* mu1 = (const float*)d_in[4];
  const float* va1 = (const float*)d_in[5];
  const float* w2  = (const float*)d_in[6];
  const float* b2  = (const float*)d_in[7];
  const float* w3  = (const float*)d_in[8];
  const float* g2  = (const float*)d_in[9];
  const float* be2 = (const float*)d_in[10];
  const float* mu2 = (const float*)d_in[11];
  const float* va2 = (const float*)d_in[12];
  const float* w4  = (const float*)d_in[13];
  const float* b4  = (const float*)d_in[14];

  float* tokens      = (float*)d_out;
  float* centers_out = tokens + (size_t)M_ * TOK_;

  char* p = (char*)d_ws;
  float* centers_ws = (float*)p;     p += (size_t)M_ * 3   * sizeof(float);
  int*   kidx       = (int*)p;       p += (size_t)M_ * K_  * sizeof(int);
  _Float16* w2h     = (_Float16*)p;  p += (size_t)256*128  * sizeof(_Float16);
  _Float16* w3h     = (_Float16*)p;  p += (size_t)512*512  * sizeof(_Float16);
  _Float16* w4h     = (_Float16*)p;  p += (size_t)384*512  * sizeof(_Float16);
  float* s1  = (float*)p;  p += 128 * sizeof(float);
  float* b1n = (float*)p;  p += 128 * sizeof(float);
  float* s2  = (float*)p;  p += 512 * sizeof(float);
  float* b2n = (float*)p;  p += 512 * sizeof(float);

  const int totalPrep = 256*128 + 512*512 + 384*512 + 128 + 512;
  prep_kernel<<<(totalPrep + 255)/256, 256, 0, stream>>>(
      w2, w3, w4, g1, be1, mu1, va1, g2, be2, mu2, va2,
      w2h, w3h, w4h, s1, b1n, s2, b2n);
  fps_kernel<<<B_, 512, 0, stream>>>(pts, centers_ws, centers_out);
  knn_kernel<<<M_, 256, 0, stream>>>(pts, centers_ws, kidx);
  mlp_kernel<<<M_ / GPB_, 256, 0, stream>>>(pts, kidx, centers_ws, w1, s1, b1n,
                                            w2h, b2, w3h, s2, b2n, w4h, b4, tokens);
  (void)in_sizes; (void)n_in; (void)out_size; (void)ws_size;
}
